// AffineCoupling_884763263531
// MI455X (gfx1250) — matrix operand order repaired and checked by reading, not compiled, not run on hardware
//
#include <hip/hip_runtime.h>
#include <cstdint>

typedef __attribute__((ext_vector_type(16))) __bf16 v16bf;
typedef __attribute__((ext_vector_type(8)))  float  v8f;

#define B_ROWS 65536
#define D_FULL 512
#define HALF   256
#define HID    2048

// ---------------- LDS layout (bytes) ----------------
#define ABUF_OFF 0
#define ABUF_SZ  (4*8*1024)            // a-tile, 4 mtiles x 8 ktiles, 1KB/tile = 32768
#define HBUF_OFF ABUF_SZ
#define HBUF_SZ  (4*2*1024)            // hidden chunk, A-layout, 8192
#define SBUF_OFF (HBUF_OFF + HBUF_SZ)  // 40960
#define SSTR 264                        // 256 + 8 pad (floats) to dodge bank conflicts
#define SBUF_SZ  (64*SSTR*4)            // 67584
#define RED_OFF  (SBUF_OFF + SBUF_SZ)   // 108544
#define RED_SZ   (256*4)
#define SMEM_SZ  (RED_OFF + RED_SZ)     // 109568 bytes < 320KB WGP budget

// A-operand (16-bit, 16x32) byte offset inside a 1KB tile for element (row r, k kin).
// ISA layout: lanes 0-15 hold K=0..7 then 16..23; lanes 16-31 hold K=8..15 then 24..31.
__device__ __forceinline__ int a_layout_off(int r, int kin) {
    int grp   = kin >> 3;                       // 0..3
    int lanep = r + ((grp & 1) << 4);           // odd groups -> lanes 16..31
    int pos   = (kin & 7) + ((grp >> 1) << 3);  // groups 2,3 -> upper half of lane's 16 elems
    return lanep * 32 + pos * 2;
}

__device__ __forceinline__ v8f wmma_bf16(v16bf a, v16bf b, v8f c) {
    // D = A(16x32 bf16) x B(32x16 bf16) + C(16x16 f32)
    return __builtin_amdgcn_wmma_f32_16x16x32_bf16(false, a, false, b, (short)0, c,
                                                   false, false);
}

__device__ __forceinline__ v8f zero8() {
    v8f z = {0.f, 0.f, 0.f, 0.f, 0.f, 0.f, 0.f, 0.f};
    return z;
}

// ---------------------------------------------------------------------------
// Weight preprocessing: f32 row-major (K x N) -> bf16 tiles in WMMA B-operand
// layout. Tile = 32(K) x 16(N) = 1KB, stored lane-major (32B per lane):
//   lanes 0-15:  column N=n, K=0..7 then 16..23 (2 bf16 per dword)
//   lanes 16-31: column N=n, K=8..15 then 24..31
// kt_major=0: tiles ordered [nt][kt] (GEMM1 walks kt inner)
// kt_major=1: tiles ordered [kt][nt] (GEMM2 walks nt inner)
// ---------------------------------------------------------------------------
__global__ void prep_w(const float* __restrict__ W, __bf16* __restrict__ dst,
                       int K, int N, int kt_major) {
    int tid = blockIdx.x * 256 + threadIdx.x;
    if (tid >= K * N) return;
    int k = tid / N, n = tid % N;
    int kt = k >> 5, kin = k & 31, nt = n >> 4, nin = n & 15;
    int lane = nin + (((kin >> 3) & 1) << 4);
    int pos  = (kin & 7) + ((kin >> 4) << 3);
    int tile = kt_major ? (kt * (N >> 4) + nt) : (nt * (K >> 5) + kt);
    dst[(size_t)tile * 512 + lane * 16 + pos] = (__bf16)W[tid];
}

// ---------------------------------------------------------------------------
// Fused affine-coupling kernel. One block = 64 rows. 8 waves.
// Per net (s then t):
//   loop over hidden in chunks of 64:
//     phase1: h = relu(a @ W1[:,chunk] + b1)      (wave: 2 WMMA x 8 ksteps)
//     phase2: acc += h @ W2[chunk,:]              (wave: 8 WMMA x 2 ksteps)
// net s: spill s -> LDS, logdet row-sums, sbuf := exp(s)
// net t: z_out = interleave(a, b*exp(s) + t) as float2 stores
// ---------------------------------------------------------------------------
__global__ __launch_bounds__(256)
void coupling_kernel(const float* __restrict__ z,
                     const __bf16* __restrict__ w1s, const float* __restrict__ b1s,
                     const __bf16* __restrict__ w2s, const float* __restrict__ b2s,
                     const __bf16* __restrict__ w1t, const float* __restrict__ b1t,
                     const __bf16* __restrict__ w2t, const float* __restrict__ b2t,
                     float* __restrict__ zout, float* __restrict__ logdet)
{
    __shared__ alignas(128) unsigned char smem[SMEM_SZ];
    const int tid  = threadIdx.x;
    const int lane = tid & 31;
    const int w    = tid >> 5;
    const int row0 = blockIdx.x * 64;

    // ---- stage a-tile (even columns of z), f32 -> bf16, into A-operand layout ----
    for (int i = tid; i < 64 * 256; i += 256) {
        int row = i >> 8, col = i & 255;
        float2 zp = *reinterpret_cast<const float2*>(z + (size_t)(row0 + row) * D_FULL + 2 * col);
        int off = ((row >> 4) * 8 + (col >> 5)) * 1024 + a_layout_off(row & 15, col & 31);
        *reinterpret_cast<__bf16*>(smem + off) = (__bf16)zp.x;
    }
    __syncthreads();

    float* sbuf = reinterpret_cast<float*>(smem + SBUF_OFF);
    float* red  = reinterpret_cast<float*>(smem + RED_OFF);

    const int m  = w >> 1;   // row tile 0..3 owned by this wave
    const int nh = w & 1;    // which half of the N range this wave owns

    for (int net = 0; net < 2; ++net) {
        const __bf16* w1 = net ? w1t : w1s;
        const float*  c1 = net ? b1t : b1s;
        const __bf16* w2 = net ? w2t : w2s;
        const float*  c2 = net ? b2t : b2s;

        v8f acc[8];
        #pragma unroll
        for (int j = 0; j < 8; ++j) acc[j] = zero8();

        #pragma unroll 1
        for (int c = 0; c < HID / 64; ++c) {
            // ---------- phase 1: hidden chunk ----------
            v8f h0 = zero8(), h1 = zero8();
            const int nt0 = c * 4 + nh * 2;   // global hidden n-tile base for this wave
            #pragma unroll
            for (int kt = 0; kt < 8; ++kt) {
                v16bf A  = *reinterpret_cast<const v16bf*>(smem + (m * 8 + kt) * 1024 + lane * 32);
                v16bf B0 = *reinterpret_cast<const v16bf*>(w1 + ((size_t)(nt0 + 0) * 8 + kt) * 512 + lane * 16);
                v16bf B1 = *reinterpret_cast<const v16bf*>(w1 + ((size_t)(nt0 + 1) * 8 + kt) * 512 + lane * 16);
                h0 = wmma_bf16(A, B0, h0);
                h1 = wmma_bf16(A, B1, h1);
            }
            // bias + relu, scatter to hbuf in A-operand layout (C-layout -> A-layout)
            {
                const int rloc = 8 * (lane >> 4);
                #pragma unroll
                for (int j = 0; j < 2; ++j) {
                    int ntl = nh * 2 + j;                        // 0..3 within chunk
                    float bb = c1[c * 64 + ntl * 16 + (lane & 15)];
                    v8f  hv  = j ? h1 : h0;
                    int  kc  = ntl * 16 + (lane & 15);           // k within chunk 0..63
                    int  tb  = HBUF_OFF + (m * 2 + (kc >> 5)) * 1024;
                    #pragma unroll
                    for (int i = 0; i < 8; ++i) {
                        float v = hv[i] + bb;
                        v = v > 0.f ? v : 0.f;
                        *reinterpret_cast<__bf16*>(smem + tb + a_layout_off(i + rloc, kc & 31)) = (__bf16)v;
                    }
                }
            }
            __syncthreads();
            // ---------- phase 2: accumulate output ----------
            #pragma unroll
            for (int kt2 = 0; kt2 < 2; ++kt2) {
                v16bf A = *reinterpret_cast<const v16bf*>(smem + HBUF_OFF + (m * 2 + kt2) * 1024 + lane * 32);
                int ktg = c * 2 + kt2;
                #pragma unroll
                for (int j = 0; j < 8; ++j) {
                    int nt = nh * 8 + j;
                    v16bf Bv = *reinterpret_cast<const v16bf*>(w2 + (size_t)(ktg * 16 + nt) * 512 + lane * 16);
                    acc[j] = wmma_bf16(A, Bv, acc[j]);
                }
            }
            __syncthreads();
        }

        if (net == 0) {
            // spill s (+ output bias) to LDS
            #pragma unroll
            for (int j = 0; j < 8; ++j) {
                int col = (nh * 8 + j) * 16 + (lane & 15);
                float bb = c2[col];
                #pragma unroll
                for (int i = 0; i < 8; ++i) {
                    int row = m * 16 + i + 8 * (lane >> 4);
                    sbuf[row * SSTR + col] = acc[j][i] + bb;
                }
            }
            __syncthreads();
            // logdet partial sums + in-place exp(s)
            {
                int row = tid >> 2, q = tid & 3;
                float sum = 0.f;
                for (int cc = 0; cc < 64; ++cc) {
                    int col = q * 64 + cc;
                    float v = sbuf[row * SSTR + col];
                    sum += v;
                    sbuf[row * SSTR + col] = __expf(v);
                }
                red[tid] = sum;
            }
            __syncthreads();
            if (tid < 64)
                logdet[row0 + tid] = red[tid*4] + red[tid*4+1] + red[tid*4+2] + red[tid*4+3];
            __syncthreads();
        } else {
            // final: z_out = interleave(a, b*exp(s) + t); 8B stores cover both columns
            #pragma unroll
            for (int j = 0; j < 8; ++j) {
                int col = (nh * 8 + j) * 16 + (lane & 15);
                float bb = c2[col];
                #pragma unroll
                for (int i = 0; i < 8; ++i) {
                    int row = m * 16 + i + 8 * (lane >> 4);
                    float tval = acc[j][i] + bb;
                    float es   = sbuf[row * SSTR + col];
                    size_t gi  = (size_t)(row0 + row) * D_FULL + 2 * col;
                    float2 zp  = *reinterpret_cast<const float2*>(z + gi);
                    float2 o;
                    o.x = zp.x;                 // a passes through
                    o.y = zp.y * es + tval;     // b * exp(s) + t
                    *reinterpret_cast<float2*>(zout + gi) = o;
                }
            }
        }
    }
}

extern "C" void kernel_launch(void* const* d_in, const int* in_sizes, int n_in,
                              void* d_out, int out_size, void* d_ws, size_t ws_size,
                              hipStream_t stream) {
    const float* z   = (const float*)d_in[0];
    const float* W1s = (const float*)d_in[1];
    const float* b1s = (const float*)d_in[2];
    const float* W2s = (const float*)d_in[3];
    const float* b2s = (const float*)d_in[4];
    const float* W1t = (const float*)d_in[5];
    const float* b1t = (const float*)d_in[6];
    const float* W2t = (const float*)d_in[7];
    const float* b2t = (const float*)d_in[8];

    // bf16 tiled weights in workspace: 4 x 524288 elems = 4MB
    __bf16* ws  = (__bf16*)d_ws;
    __bf16* w1s = ws;
    __bf16* w1t = ws + (size_t)524288;
    __bf16* w2s = ws + (size_t)2 * 524288;
    __bf16* w2t = ws + (size_t)3 * 524288;

    const int nel = HALF * HID;          // 524288
    const int pb  = (nel + 255) / 256;
    prep_w<<<pb, 256, 0, stream>>>(W1s, w1s, HALF, HID, 0);
    prep_w<<<pb, 256, 0, stream>>>(W1t, w1t, HALF, HID, 0);
    prep_w<<<pb, 256, 0, stream>>>(W2s, w2s, HID, HALF, 1);
    prep_w<<<pb, 256, 0, stream>>>(W2t, w2t, HID, HALF, 1);

    float* zout   = (float*)d_out;
    float* logdet = zout + (size_t)B_ROWS * D_FULL;
    coupling_kernel<<<B_ROWS / 64, 256, 0, stream>>>(z, w1s, b1s, w2s, b2s,
                                                     w1t, b1t, w2t, b2t,
                                                     zout, logdet);
}
